// Conv2d_87239375716489
// MI455X (gfx1250) — matrix operand order repaired and checked by reading, not compiled, not run on hardware
//
#include <hip/hip_runtime.h>

// CDNA5 / gfx1250 implicit-GEMM 3x3 conv (padding=1), bf16 WMMA, f32 accumulate.
// Pass 1: pack weights fp32 OIHW -> bf16 in WMMA A-fragment order (L2-resident).
// Pass 2: GEMM; A K-slices staged once per block into LDS (shared by all waves),
//         B via im2col gather -> LDS; both double-buffered.

typedef __bf16 bf16_t;
typedef __attribute__((ext_vector_type(16))) __bf16 v16bf;
typedef __attribute__((ext_vector_type(8)))  __bf16 v8bf;
typedef __attribute__((ext_vector_type(8)))  float   v8f;

union V16U { v16bf v; v8bf h[2]; };

#define CIN 256
#define COUT 256
#define IMW 112
#define HW (112 * 112)          // 12544
#define KB 32                   // K per WMMA step
#define KSTEPS 72               // 9 taps * 8 chunks of 32 channels
#define BN 64                   // spatial positions per block
#define BPITCH 40               // bf16 per LDS row (32 + pad)
#define MT_TOT 16               // 16 m-tiles of 16 output channels
#define FRAG_ELEMS 512          // 32 lanes * 16 bf16 per (mt,kk) fragment block

// ---------------------------------------------------------------------------
// Pass 1: W[co][ci][ky][kx] fp32  ->  apack[mt][kk][lane][j] bf16
// K reordered as k = tap*256 + ci ; fragment element j <-> K = (j&8?16:0)+half*8+(j&7)
// ---------------------------------------------------------------------------
__global__ __launch_bounds__(256)
void pack_weights_bf16(const float* __restrict__ Wt, bf16_t* __restrict__ apack)
{
    int gid  = blockIdx.x * 256 + threadIdx.x;     // 0 .. 16*72*32*16-1
    int j    = gid & 15;
    int L    = (gid >> 4) & 31;
    int blk  = gid >> 9;                           // mt*72 + kk
    int kk   = blk % KSTEPS;
    int mt   = blk / KSTEPS;

    int m    = mt * 16 + (L & 15);
    int half = L >> 4;
    int K    = kk * KB + ((j & 8) ? 16 : 0) + half * 8 + (j & 7);
    int tap  = K >> 8;                             // 0..8  (ky*3+kx)
    int ci   = K & 255;

    apack[gid] = (bf16_t)Wt[m * (CIN * 9) + ci * 9 + tap];
}

// ---------------------------------------------------------------------------
// Pass 2: GEMM. M = 256 (all co per block), N tile = 64, K = 2304.
// 8 waves: waveM 0..3 (64 rows each, 4 m-tiles), waveN 0..1 (32 cols, 2 n-tiles).
// ---------------------------------------------------------------------------
__global__ __launch_bounds__(256)
void conv3x3_gemm_wmma(const float* __restrict__ inp,
                       const bf16_t* __restrict__ apack,
                       const float* __restrict__ bias,
                       float* __restrict__ out)
{
    __shared__ __align__(16) bf16_t Asl[2][MT_TOT * FRAG_ELEMS]; // 32 KB dbl-buf
    __shared__ __align__(16) bf16_t Bs[2][BN * BPITCH];          // 10 KB dbl-buf

    const int tid   = threadIdx.x;
    const int lane  = tid & 31;
    const int wave  = tid >> 5;
    const int half  = lane >> 4;
    const int lrow  = lane & 15;
    const int waveM = wave & 3;
    const int waveN = wave >> 2;
    const int n0    = blockIdx.x * BN;

    // ---- A staging: 16 KB K-slice per step, 4 x 16B per thread, coalesced ----
    int a_goff[4], a_loff[4];
#pragma unroll
    for (int c = 0; c < 4; ++c) {
        int linear = c * 256 + tid;            // 0..1023 (16 mt * 64 chunks)
        int mt     = linear >> 6;
        a_goff[c]  = mt * (KSTEPS * FRAG_ELEMS) + (linear & 63) * 8;
        a_loff[c]  = linear * 8;
    }
    v8bf a_reg[4];
    auto gather_a = [&](int kk) {
#pragma unroll
        for (int c = 0; c < 4; ++c)
            a_reg[c] = *(const v8bf*)(apack + a_goff[c] + kk * FRAG_ELEMS);
        int kpf = (kk + 1 < KSTEPS) ? kk + 1 : kk;
        __builtin_prefetch(apack + a_goff[0] + kpf * FRAG_ELEMS, 0, 3);
    };
    auto store_a = [&](int buf) {
#pragma unroll
        for (int c = 0; c < 4; ++c)
            *(v8bf*)(&Asl[buf][a_loff[c]]) = a_reg[c];
    };

    // ---- B gather descriptors: 8 elements/thread/step, coalesced across lanes ----
    int b_nloc[8], b_kloc[8], b_loff[8], b_addr[8];
    float b_okf[8];
#pragma unroll
    for (int i = 0; i < 8; ++i) {
        int idx   = i * 256 + tid;
        b_kloc[i] = idx >> 6;        // 0..31 within K chunk (= ci low bits)
        b_nloc[i] = idx & 63;        // consecutive tid -> consecutive pos
        b_loff[i] = b_nloc[i] * BPITCH + b_kloc[i];
    }

    // Per-tap setup: (yy,xx,valid) fixed for 8 consecutive K-steps. Invalid taps
    // get a clamped in-bounds address; zeroing is a multiply by 0.0f at store.
    auto setup_tap = [&](int tap) {
        int ky = tap / 3, kx = tap - ky * 3;
#pragma unroll
        for (int i = 0; i < 8; ++i) {
            int pos = n0 + b_nloc[i];
            int nb  = pos / HW;
            int rem = pos - nb * HW;
            int y   = rem / IMW;
            int x   = rem - y * IMW;
            int yy  = y + ky - 1;
            int xx  = x + kx - 1;
            int ok  = (((unsigned)yy < IMW) && ((unsigned)xx < IMW)) ? 1 : 0;
            int a   = nb * (CIN * HW) + b_kloc[i] * HW + yy * IMW + xx;
            b_okf[i]  = ok ? 1.0f : 0.0f;
            b_addr[i] = ok ? a : 0;      // stays in-bounds across the whole tap
        }
    };

    float b_reg[8];
    auto gather_b = [&]() {              // 8 unconditional loads (one clause)
#pragma unroll
        for (int i = 0; i < 8; ++i) {
            b_reg[i]   = inp[b_addr[i]];
            b_addr[i] += KB * HW;
        }
    };
    auto store_b = [&](int buf) {        // mask (x0.0/x1.0) + convert at store
#pragma unroll
        for (int i = 0; i < 8; ++i)
            Bs[buf][b_loff[i]] = (bf16_t)(b_reg[i] * b_okf[i]);
    };

    const v8f vzero = {0.f, 0.f, 0.f, 0.f, 0.f, 0.f, 0.f, 0.f};
    v8f acc[4][2];
#pragma unroll
    for (int m = 0; m < 4; ++m)
#pragma unroll
        for (int n = 0; n < 2; ++n) acc[m][n] = vzero;

    auto compute = [&](int buf) {
        // B fragments from LDS: lane = n, K = half*8 + 0..7, then 16 + half*8 + 0..7
        v16bf bfrag[2];
#pragma unroll
        for (int nt = 0; nt < 2; ++nt) {
            const bf16_t* p =
                &Bs[buf][(waveN * 32 + nt * 16 + lrow) * BPITCH + half * 8];
            V16U u;
            u.h[0] = *(const v8bf*)(p);
            u.h[1] = *(const v8bf*)(p + 16);
            bfrag[nt] = u.v;
        }
        // A fragments from the block-shared LDS K-slice
        v16bf afrag[4];
#pragma unroll
        for (int mti = 0; mti < 4; ++mti) {
            const bf16_t* p =
                &Asl[buf][(waveM * 4 + mti) * FRAG_ELEMS + lane * 16];
            V16U u;
            u.h[0] = *(const v8bf*)(p);
            u.h[1] = *(const v8bf*)(p + 8);
            afrag[mti] = u.v;
        }
#pragma unroll
        for (int mti = 0; mti < 4; ++mti)
#pragma unroll
            for (int nt = 0; nt < 2; ++nt)
                acc[mti][nt] = __builtin_amdgcn_wmma_f32_16x16x32_bf16(
                    false, afrag[mti], false, bfrag[nt],
                    (short)0, acc[mti][nt], false, false);
    };

    // ---- software-pipelined main loop (A and B double-buffered in LDS) ----
    setup_tap(0);
    gather_b();
    gather_a(0);
    store_b(0);
    store_a(0);
    __syncthreads();

    int buf = 0;
    for (int tap = 0; tap < 9; ++tap) {
#pragma unroll
        for (int c = 0; c < 8; ++c) {
            int kk = tap * 8 + c;
            bool last = (kk == KSTEPS - 1);
            if (!last) {
                if (c == 7) setup_tap(tap + 1);   // resolved at compile time
                gather_b();
                gather_a(kk + 1);
            }
            compute(buf);
            __syncthreads();
            if (!last) { store_b(buf ^ 1); store_a(buf ^ 1); }
            __syncthreads();
            buf ^= 1;
        }
    }

    // ---- epilogue: C/D layout -> NCHW (+bias), 16-lane contiguous in x ----
#pragma unroll
    for (int nt = 0; nt < 2; ++nt) {
        int pos = n0 + waveN * 32 + nt * 16 + lrow;
        int nb  = pos / HW;
        int rem = pos - nb * HW;
        int y   = rem / IMW;
        int x   = rem - y * IMW;
        float* obase = out + (size_t)nb * COUT * HW + y * IMW + x;
#pragma unroll
        for (int mti = 0; mti < 4; ++mti) {
            int co0 = waveM * 64 + mti * 16 + half * 8;
#pragma unroll
            for (int r = 0; r < 8; ++r) {
                int co = co0 + r;
                obase[(size_t)co * HW] = acc[mti][nt][r] + bias[co];
            }
        }
    }
}

// ---------------------------------------------------------------------------
extern "C" void kernel_launch(void* const* d_in, const int* in_sizes, int n_in,
                              void* d_out, int out_size, void* d_ws, size_t ws_size,
                              hipStream_t stream) {
    (void)in_sizes; (void)n_in; (void)out_size; (void)ws_size;
    const float* inp  = (const float*)d_in[0];   // (16,256,112,112) f32
    const float* Wt   = (const float*)d_in[1];   // (256,256,3,3)    f32
    const float* bias = (const float*)d_in[2];   // (256,1,1)        f32
    float* out        = (float*)d_out;           // (16,256,112,112) f32
    bf16_t* apack     = (bf16_t*)d_ws;           // 16*72*512 bf16 = 1.13 MB

    // Pass 1: pack weights (589824 elements)
    const int packN = MT_TOT * KSTEPS * FRAG_ELEMS;          // 589824
    pack_weights_bf16<<<packN / 256, 256, 0, stream>>>(Wt, apack);

    // Pass 2: GEMM over all spatial positions
    const int Ntot = 16 * HW;                                // 200704
    conv3x3_gemm_wmma<<<Ntot / BN, 256, 0, stream>>>(inp, apack, bias, out);
}
